// TFGPTJAttention_84859963834465
// MI455X (gfx1250) — matrix operand order repaired and checked by reading, not compiled, not run on hardware
//
#include <hip/hip_runtime.h>
#include <hip/hip_bf16.h>
#include <math.h>

typedef __attribute__((ext_vector_type(16))) _Float16 v16h;
typedef __attribute__((ext_vector_type(8)))  float    v8f;

#define S_LEN 2048
#define EMB   4096
#define NH    16
#define HD    256
#define ROTD  64

// ---------------------------------------------------------------------------
// WMMA fragment helpers (CDNA5 16x16x32 f16, wave32 layouts per ISA 7.12.2)
// A (16x32, MxK): lane m=l&15 holds row m; lanes 0-15 K={0..7,16..23},
//                 lanes 16-31 K={8..15,24..31} (two 8-f16 chunks).
// B (32x16, KxN): lane n=l&15 holds col n; lanes 0-15 K={0..7,16..23}, 16-31 K={8..15,24..31}.
//                 Source must be [N][K] contiguous (i.e. B transposed).
// C/D (16x16):    vgpr i, lanes 0-15 -> M=i, lanes 16-31 -> M=i+8; N=l&15.
// ---------------------------------------------------------------------------
__device__ __forceinline__ v16h frag_a_lds(const _Float16* base, int ld, int lane) {
    int m  = lane & 15;
    int hi = (lane >> 4) & 1;
    const _Float16* p = base + m * ld + hi * 8;
    v16h r;
#pragma unroll
    for (int i = 0; i < 8; ++i) r[i] = p[i];
#pragma unroll
    for (int i = 0; i < 8; ++i) r[8 + i] = p[16 + i];
    return r;
}

__device__ __forceinline__ v16h frag_b_lds(const _Float16* baseT, int ld, int lane) {
    int n  = lane & 15;
    int hi = (lane >> 4) & 1;
    const _Float16* p = baseT + n * ld + hi * 8;
    v16h r;
#pragma unroll
    for (int i = 0; i < 8; ++i) r[i] = p[i];
#pragma unroll
    for (int i = 0; i < 8; ++i) r[8 + i] = p[16 + i];
    return r;
}

// ---------------------------------------------------------------------------
// f32 -> f16 cast
// ---------------------------------------------------------------------------
__global__ void cast_f32_f16(const float* __restrict__ in, _Float16* __restrict__ out, int n) {
    int idx = (blockIdx.x * blockDim.x + threadIdx.x) * 4;
    if (idx >= n) return;
    float4 v = *(const float4*)(in + idx);
    out[idx + 0] = (_Float16)v.x;
    out[idx + 1] = (_Float16)v.y;
    out[idx + 2] = (_Float16)v.z;
    out[idx + 3] = (_Float16)v.w;
}

// ---------------------------------------------------------------------------
// C[M][N] = X[M][K](f16) @ W[K][N](f32, converted to f16 on the fly)
// M=2048, N=K=4096. Block: 256 thr = 8 waves (4 along M x 2 along N),
// tile 128x128x32, double-buffered LDS so tile k+1 global loads overlap the
// WMMAs of tile k.
// ---------------------------------------------------------------------------
template <bool F32OUT>
__global__ __launch_bounds__(256, 1) void gemm_xw(const _Float16* __restrict__ X,
                                                  const float* __restrict__ W,
                                                  void* __restrict__ Cout) {
    const int N = EMB, K = EMB;
    __shared__ _Float16 Al[2][128 * 40];   // [row][k], ld=40
    __shared__ _Float16 Bt[2][128 * 40];   // [n][k],  ld=40
    int tid  = threadIdx.x;
    int lane = tid & 31, wid = tid >> 5;
    int n0 = blockIdx.x * 128;
    int m0 = blockIdx.y * 128;
    int wm = (wid >> 1) * 32;   // wave row offset (4 waves along M)
    int wn = (wid & 1) * 64;    // wave col offset (2 waves along N)

    auto load_tiles = [&](int k0, _Float16* Abuf, _Float16* Bbuf) {
        // A tile: 128x32 f16, 512 chunks of 8 f16, 2 per thread
#pragma unroll
        for (int i = 0; i < 2; ++i) {
            int idx = tid + i * 256;
            int row = idx >> 2;
            int ch  = idx & 3;
            const _Float16* src = X + (size_t)(m0 + row) * K + k0 + ch * 8;
            *(uint4*)&Abuf[row * 40 + ch * 8] = *(const uint4*)src;
        }
        // W tile: 32x128 f32 -> f16, transposed into Bt[n][k]
#pragma unroll
        for (int i = 0; i < 4; ++i) {
            int idx = tid + i * 256;        // 1024 float4 groups
            int kk  = (idx * 4) >> 7;       // 0..31
            int nn  = (idx * 4) & 127;
            float4 wv = *(const float4*)(W + (size_t)(k0 + kk) * N + n0 + nn);
            Bbuf[(nn + 0) * 40 + kk] = (_Float16)wv.x;
            Bbuf[(nn + 1) * 40 + kk] = (_Float16)wv.y;
            Bbuf[(nn + 2) * 40 + kk] = (_Float16)wv.z;
            Bbuf[(nn + 3) * 40 + kk] = (_Float16)wv.w;
        }
    };

    v8f acc[2][4] = {};

    load_tiles(0, Al[0], Bt[0]);
    __syncthreads();

    int buf = 0;
    for (int k0 = 0; k0 < K; k0 += 32, buf ^= 1) {
        if (k0 + 32 < K) load_tiles(k0 + 32, Al[buf ^ 1], Bt[buf ^ 1]);

        v16h af[2], bf[4];
#pragma unroll
        for (int a = 0; a < 2; ++a) af[a] = frag_a_lds(&Al[buf][(wm + a * 16) * 40], 40, lane);
#pragma unroll
        for (int b = 0; b < 4; ++b) bf[b] = frag_b_lds(&Bt[buf][(wn + b * 16) * 40], 40, lane);
#pragma unroll
        for (int a = 0; a < 2; ++a)
#pragma unroll
            for (int b = 0; b < 4; ++b)
                acc[a][b] = __builtin_amdgcn_wmma_f32_16x16x32_f16(
                    false, af[a], false, bf[b], (short)0, acc[a][b], false, false);
        __syncthreads();
    }

    int nloc = lane & 15;
    int mo   = (lane >> 4) * 8;
#pragma unroll
    for (int a = 0; a < 2; ++a)
#pragma unroll
        for (int b = 0; b < 4; ++b) {
            int mbase = m0 + wm + a * 16 + mo;
            int nbase = n0 + wn + b * 16 + nloc;
#pragma unroll
            for (int i = 0; i < 8; ++i) {
                float  val = acc[a][b][i];
                size_t off = (size_t)(mbase + i) * N + nbase;
                if (F32OUT) ((float*)Cout)[off] = val;
                else        ((_Float16*)Cout)[off] = (_Float16)val;
            }
        }
}

// ---------------------------------------------------------------------------
// GPT-J rotary (reference variant: r = (x[2i+1], x[2i]), NO negation),
// applied in place to the first ROTD dims of every head of Q and K.
// One thread per (s, h, freq-pair).
// ---------------------------------------------------------------------------
__global__ void rotary_kernel(_Float16* __restrict__ Qh, _Float16* __restrict__ Kh) {
    int idx = blockIdx.x * blockDim.x + threadIdx.x;
    int i = idx & 31;          // freq index 0..31
    int h = (idx >> 5) & 15;
    int s = idx >> 9;
    if (s >= S_LEN) return;
    float inv = __powf(10000.0f, -(float)(2 * i) * (1.0f / (float)ROTD));
    float ang = (float)s * inv;
    float sn, cs;
    __sincosf(ang, &sn, &cs);
    size_t base = (size_t)s * EMB + (size_t)h * HD + 2 * i;
    float q0 = (float)Qh[base], q1 = (float)Qh[base + 1];
    Qh[base]     = (_Float16)(q0 * cs + q1 * sn);
    Qh[base + 1] = (_Float16)(q1 * cs + q0 * sn);
    float k0 = (float)Kh[base], k1 = (float)Kh[base + 1];
    Kh[base]     = (_Float16)(k0 * cs + k1 * sn);
    Kh[base + 1] = (_Float16)(k1 * cs + k0 * sn);
}

// ---------------------------------------------------------------------------
// Flash attention, causal, mask-before-scale (GPT-J quirk).
// Grid: (S/128, H). Block: 256 thr = 8 waves; wave w owns queries
// q0 = qt*128 + w*16 .. +15. Q A-frags are RE-LOADED from global (L2-hot)
// every key tile to keep VGPR pressure low (no spills); O accumulators
// (16 x v8f = 128 VGPRs) stay in registers.
// ---------------------------------------------------------------------------
__global__ __launch_bounds__(256, 1) void attn_kernel(const _Float16* __restrict__ Q,
                                                      const _Float16* __restrict__ Km,
                                                      const _Float16* __restrict__ V,
                                                      _Float16* __restrict__ O) {
    __shared__ _Float16 Kl[32 * 264];       // [key][d],  ld=264
    __shared__ _Float16 Vt[256 * 40];       // [d][key],  ld=40
    __shared__ _Float16 Pl[8][16 * 40];     // per-wave P tile [q][key], ld=40

    int tid = threadIdx.x, lane = tid & 31, wid = tid >> 5;
    int qt = blockIdx.x, h = blockIdx.y;
    int q0 = qt * 128 + wid * 16;
    int m = lane & 15, hi = lane >> 4, nloc = m;
    const size_t hoff = (size_t)h * HD;
    const _Float16* qrow = Q + (size_t)(q0 + m) * EMB + hoff + hi * 8;

    v8f acc[16] = {};               // O accum: 16 d-tiles of 16x16
    float mrow[8], lrow[8];
#pragma unroll
    for (int i = 0; i < 8; ++i) { mrow[i] = -INFINITY; lrow[i] = 0.0f; }

    int nkt = (qt * 128 + 127) / 32 + 1;    // causal: key tiles needed by this block
    for (int kt = 0; kt < nkt; ++kt) {
        int kbase = kt * 32;
        // cooperative load: K tile row-major, V tile transposed into Vt
#pragma unroll
        for (int i = 0; i < 4; ++i) {
            int idx = tid + i * 256;
            int row = idx >> 5;     // key 0..31
            int ch  = idx & 31;     // 8-f16 chunk of D
            const _Float16* ks = Km + (size_t)(kbase + row) * EMB + hoff + ch * 8;
            *(uint4*)&Kl[row * 264 + ch * 8] = *(const uint4*)ks;
            const _Float16* vs = V + (size_t)(kbase + row) * EMB + hoff + ch * 8;
            uint4 vv = *(const uint4*)vs;
            const _Float16* ve = (const _Float16*)&vv;
#pragma unroll
            for (int j = 0; j < 8; ++j) Vt[(ch * 8 + j) * 40 + row] = ve[j];
        }
        __syncthreads();

        // logits: 16 queries x 32 keys, K-dim = 256 (Q frags reloaded from global)
        v8f lg[2] = {};
#pragma unroll
        for (int db = 0; db < 8; ++db) {
            const _Float16* p = qrow + db * 32;
            v16h qa;
#pragma unroll
            for (int i = 0; i < 8; ++i) qa[i] = p[i];
#pragma unroll
            for (int i = 0; i < 8; ++i) qa[8 + i] = p[16 + i];
#pragma unroll
            for (int nb = 0; nb < 2; ++nb) {
                v16h bf = frag_b_lds(&Kl[(nb * 16) * 264 + db * 32], 264, lane);
                lg[nb] = __builtin_amdgcn_wmma_f32_16x16x32_f16(
                    false, qa, false, bf, (short)0, lg[nb], false, false);
            }
        }

        // mask (before scale!), scale, online softmax; lg is overwritten with exp()
        float nscale[8];
#pragma unroll
        for (int i = 0; i < 8; ++i) {
            int qg = q0 + i + 8 * hi;
#pragma unroll
            for (int nb = 0; nb < 2; ++nb) {
                int kg = kbase + nb * 16 + nloc;
                float x = lg[nb][i];
                x = (kg <= qg) ? x : -1.0e9f;
                lg[nb][i] = x * 0.0625f;    // 1/sqrt(256)
            }
            float t = fmaxf(lg[0][i], lg[1][i]);
#pragma unroll
            for (int off = 8; off >= 1; off >>= 1) t = fmaxf(t, __shfl_xor(t, off, 16));
            float mn = fmaxf(mrow[i], t);
            float sc = __expf(mrow[i] - mn);
            mrow[i] = mn;
            float s0 = 0.0f;
#pragma unroll
            for (int nb = 0; nb < 2; ++nb) {
                float e = __expf(lg[nb][i] - mn);
                lg[nb][i] = e;              // reuse lg as P
                s0 += e;
            }
#pragma unroll
            for (int off = 8; off >= 1; off >>= 1) s0 += __shfl_xor(s0, off, 16);
            lrow[i] = lrow[i] * sc + s0;
            nscale[i] = sc;
        }
#pragma unroll
        for (int t2 = 0; t2 < 16; ++t2)
#pragma unroll
            for (int i = 0; i < 8; ++i) acc[t2][i] *= nscale[i];

        // D-frag -> A-frag reshape of P via per-wave LDS (same-wave DS ops are in-order)
#pragma unroll
        for (int nb = 0; nb < 2; ++nb)
#pragma unroll
            for (int i = 0; i < 8; ++i)
                Pl[wid][(i + 8 * hi) * 40 + nb * 16 + nloc] = (_Float16)lg[nb][i];
        v16h pf = frag_a_lds(&Pl[wid][0], 40, lane);

        // O += P(16x32) @ V(32x256)
#pragma unroll
        for (int dt = 0; dt < 16; ++dt) {
            v16h vb = frag_b_lds(&Vt[(dt * 16) * 40], 40, lane);
            acc[dt] = __builtin_amdgcn_wmma_f32_16x16x32_f16(
                false, pf, false, vb, (short)0, acc[dt], false, false);
        }
        __syncthreads();
    }

    // normalize and store f16 attn-out [s][h*256+d]
#pragma unroll
    for (int i = 0; i < 8; ++i) {
        float inv = 1.0f / lrow[i];
        int qg = q0 + i + 8 * hi;
#pragma unroll
        for (int dt = 0; dt < 16; ++dt)
            O[(size_t)qg * EMB + hoff + dt * 16 + nloc] = (_Float16)(acc[dt][i] * inv);
    }
}

// ---------------------------------------------------------------------------
// Launch: cast -> QKV GEMMs -> rotary -> flash attention -> output GEMM
// Workspace: Xh,Qh,Kh,Vh,Ah f16 buffers (5 x 16 MB = 80 MB).
// ---------------------------------------------------------------------------
extern "C" void kernel_launch(void* const* d_in, const int* in_sizes, int n_in,
                              void* d_out, int out_size, void* d_ws, size_t ws_size,
                              hipStream_t stream) {
    const float* X  = (const float*)d_in[0];
    const float* Wq = (const float*)d_in[1];
    const float* Wk = (const float*)d_in[2];
    const float* Wv = (const float*)d_in[3];
    const float* Wo = (const float*)d_in[4];

    const size_t NE = (size_t)S_LEN * EMB;
    _Float16* Xh = (_Float16*)d_ws;
    _Float16* Qh = Xh + NE;
    _Float16* Kh = Qh + NE;
    _Float16* Vh = Kh + NE;
    _Float16* Ah = Vh + NE;

    cast_f32_f16<<<(int)(NE / 1024), 256, 0, stream>>>(X, Xh, (int)NE);

    dim3 gg(EMB / 128, S_LEN / 128);
    gemm_xw<false><<<gg, 256, 0, stream>>>(Xh, Wq, Qh);
    gemm_xw<false><<<gg, 256, 0, stream>>>(Xh, Wk, Kh);
    gemm_xw<false><<<gg, 256, 0, stream>>>(Xh, Wv, Vh);

    rotary_kernel<<<(S_LEN * NH * (ROTD / 2)) / 256, 256, 0, stream>>>(Qh, Kh);

    attn_kernel<<<dim3(S_LEN / 128, NH), 256, 0, stream>>>(Qh, Kh, Vh, Ah);

    gemm_xw<true><<<gg, 256, 0, stream>>>(Ah, Wo, d_out);
}
